// SparseSpikeFullAttention_61306363183704
// MI455X (gfx1250) — matrix operand order repaired and checked by reading, not compiled, not run on hardware
//
#include <hip/hip_runtime.h>

typedef __attribute__((ext_vector_type(16))) _Float16 v16h;
typedef __attribute__((ext_vector_type(8)))  _Float16 v8h;
typedef __attribute__((ext_vector_type(8)))  float    v8f;

#define B_  2
#define T_  8
#define N_  1024
#define D_  512
#define H_  8
#define DH_ 64
#define FR_ 32
#define ROWS_ (B_*T_*N_)          // 16384
#define BTH_  (B_*T_*H_)          // 128
static constexpr float RMS_EPS = 1e-6f;
static constexpr float NEGV    = -1e30f;

// ---------------------------------------------------------------- helpers ----
union V16u { v16h v; v8h h[2]; };

__device__ __forceinline__ v16h ld16(const _Float16* p0, const _Float16* p1) {
  V16u u;
  u.h[0] = *(const v8h*)p0;
  u.h[1] = *(const v8h*)p1;
  return u.v;
}

__device__ __forceinline__ v8f wmma32(v16h a, v16h b, v8f c) {
  // D = A(16x32 f16) * B(32x16 f16) + C(16x16 f32)
  return __builtin_amdgcn_wmma_f32_16x16x32_f16(false, a, false, b, (short)0, c,
                                                false, false);
}

// reductions within a 16-lane half-wave (rows of one C-tile half)
__device__ __forceinline__ float rmax16(float v) {
  v = fmaxf(v, __shfl_xor(v, 1, 16));
  v = fmaxf(v, __shfl_xor(v, 2, 16));
  v = fmaxf(v, __shfl_xor(v, 4, 16));
  v = fmaxf(v, __shfl_xor(v, 8, 16));
  return v;
}
__device__ __forceinline__ float rsum16(float v) {
  v += __shfl_xor(v, 1, 16);
  v += __shfl_xor(v, 2, 16);
  v += __shfl_xor(v, 4, 16);
  v += __shfl_xor(v, 8, 16);
  return v;
}

// ---------------------------------------------------------------- kernels ----
// RoPE table: rope[b,n,0:32]=sin(ang), rope[b,n,32:64]=cos(ang)
__global__ void k_rope(const float* __restrict__ pos, const float* __restrict__ dirs,
                       const float* __restrict__ freqs, float* __restrict__ rope) {
  int tid = blockIdx.x * blockDim.x + threadIdx.x;   // B_*N_*FR_ = 65536
  if (tid >= B_ * N_ * FR_) return;
  int f  = tid & (FR_ - 1);
  int bn = tid >> 5;
  float px = pos[bn * 3 + 0], py = pos[bn * 3 + 1], pz = pos[bn * 3 + 2];
  float nrm = fmaxf(sqrtf(px * px + py * py + pz * pz), 1e-12f);
  float inv = 1.0f / nrm;
  float ang = (px * dirs[f * 3] + py * dirs[f * 3 + 1] + pz * dirs[f * 3 + 2]) * inv *
              freqs[f];
  rope[(size_t)bn * 64 + f]      = sinf(ang);
  rope[(size_t)bn * 64 + 32 + f] = cosf(ang);
}

// key mask: send[b,t,n] = spike && keep ; anySend[b,t] = OR over n
__global__ void k_send(const int* __restrict__ pad, const int* __restrict__ spike,
                       unsigned char* __restrict__ send, int* __restrict__ anySend) {
  int bt = blockIdx.x;          // 0..15
  int b  = bt >> 3;             // T_=8
  __shared__ int any;
  if (threadIdx.x == 0) any = 0;
  __syncthreads();
  int local = 0;
  for (int n = threadIdx.x; n < N_; n += blockDim.x) {
    int s = (spike[(size_t)bt * N_ + n] != 0) && (pad[(size_t)b * N_ + n] != 0);
    send[(size_t)bt * N_ + n] = (unsigned char)s;
    local |= s;
  }
  if (local) atomicOr(&any, 1);
  __syncthreads();
  if (threadIdx.x == 0) anySend[bt] = any;
}

// weights fp32 -> f16 : w16 = [Wq16 | Wk16 | Wv16 | Wo16]
__global__ void k_wconv(const float* __restrict__ wq, const float* __restrict__ wk,
                        const float* __restrict__ wv, const float* __restrict__ wo,
                        _Float16* __restrict__ w16) {
  int tid   = blockIdx.x * blockDim.x + threadIdx.x;  // 4*262144
  int which = tid >> 18;
  int off   = tid & ((1 << 18) - 1);
  const float* s;
  if (which == 0) s = wq;
  else if (which == 1) s = wk;
  else if (which == 2) s = wv;
  else s = wo;
  w16[tid] = (_Float16)s[off];
}

// RMSNorm + rope add; one block per row of D_=512
__global__ __launch_bounds__(256) void k_norm(const float* __restrict__ x,
                                              const float* __restrict__ rms_w,
                                              const float* __restrict__ rope,
                                              _Float16* __restrict__ xn16,
                                              _Float16* __restrict__ qk16) {
  int r   = blockIdx.x;           // 0..16383
  int tid = threadIdx.x;          // 256
  const float* xr = x + (size_t)r * D_;
  float v0 = xr[tid], v1 = xr[tid + 256];
  float ss = v0 * v0 + v1 * v1;
  ss += __shfl_xor(ss, 16, 32);
  ss += __shfl_xor(ss, 8, 32);
  ss += __shfl_xor(ss, 4, 32);
  ss += __shfl_xor(ss, 2, 32);
  ss += __shfl_xor(ss, 1, 32);
  __shared__ float red[8];
  if ((tid & 31) == 0) red[tid >> 5] = ss;
  __syncthreads();
  float tot = red[0] + red[1] + red[2] + red[3] + red[4] + red[5] + red[6] + red[7];
  float sc = rsqrtf(tot * (1.0f / (float)D_) + RMS_EPS);

  int n = r & (N_ - 1);
  int b = r >> 13;                // r / (T_*N_)
  const float* rp = rope + ((size_t)b * N_ + n) * 64;

  {
    int   d  = tid;
    float xv = v0 * sc * rms_w[d];
    xn16[(size_t)r * D_ + d] = (_Float16)xv;
    float qv = xv + (d < 64 ? rp[d] : 0.0f);
    qk16[(size_t)r * D_ + d] = (_Float16)qv;
  }
  {
    int   d  = tid + 256;
    float xv = v1 * sc * rms_w[d];
    xn16[(size_t)r * D_ + d] = (_Float16)xv;
    qk16[(size_t)r * D_ + d] = (_Float16)xv;   // d >= 64 always here
  }
}

// projections with 2x4 register blocking: each wave computes 32 rows x 64 cols.
// mode 0=q (qk16*WqT), 1=k (qk16*WkT), 2=v (xn16*WvT, stored transposed)
// 64-wide col block == exactly one head (h = ct).
__global__ __launch_bounds__(256) void k_proj(const _Float16* __restrict__ qk16,
                                              const _Float16* __restrict__ xn16,
                                              const _Float16* __restrict__ w16,
                                              _Float16* __restrict__ q16,
                                              _Float16* __restrict__ k16,
                                              _Float16* __restrict__ vT16) {
  int wave = threadIdx.x >> 5, lane = threadIdx.x & 31;
  int gw   = blockIdx.x * 8 + wave;       // 3*4096 waves
  int mode = gw >> 12;
  int rem  = gw & 4095;
  int rt   = rem >> 3;                    // 0..511  : rows rt*32
  int ct   = rem & 7;                     // 0..7    : cols ct*64 (== head ct)
  int ln   = lane & 15, hi = lane >> 4;

  const _Float16* in = (mode < 2) ? qk16 : xn16;
  const _Float16* W  = w16 + (size_t)mode * (D_ * D_);
  const _Float16* ar0 = in + (size_t)(rt * 32 + ln) * D_ + hi * 8;
  const _Float16* ar1 = ar0 + (size_t)16 * D_;
  const _Float16* br0 = W + (size_t)(ct * 64 + ln) * D_ + hi * 8;
  const _Float16* br1 = br0 + (size_t)16 * D_;
  const _Float16* br2 = br0 + (size_t)32 * D_;
  const _Float16* br3 = br0 + (size_t)48 * D_;

  v8f acc[2][4] = {};
#pragma unroll
  for (int kk = 0; kk < 16; ++kk) {
    int kb = kk * 32;
    v16h a0 = ld16(ar0 + kb, ar0 + kb + 16);
    v16h a1 = ld16(ar1 + kb, ar1 + kb + 16);
    v16h b0 = ld16(br0 + kb, br0 + kb + 16);
    v16h b1 = ld16(br1 + kb, br1 + kb + 16);
    v16h b2 = ld16(br2 + kb, br2 + kb + 16);
    v16h b3 = ld16(br3 + kb, br3 + kb + 16);
    acc[0][0] = wmma32(a0, b0, acc[0][0]);
    acc[0][1] = wmma32(a0, b1, acc[0][1]);
    acc[0][2] = wmma32(a0, b2, acc[0][2]);
    acc[0][3] = wmma32(a0, b3, acc[0][3]);
    acc[1][0] = wmma32(a1, b0, acc[1][0]);
    acc[1][1] = wmma32(a1, b1, acc[1][1]);
    acc[1][2] = wmma32(a1, b2, acc[1][2]);
    acc[1][3] = wmma32(a1, b3, acc[1][3]);
  }

  int h = ct;                              // col block == head
#pragma unroll
  for (int i = 0; i < 2; ++i) {
    int rbase = rt * 32 + i * 16 + hi * 8; // global row of first held element
    int bt = rbase >> 10;                  // tiles never straddle bt (1024%32==0)
    int nb = rbase & (N_ - 1);
#pragma unroll
    for (int j = 0; j < 4; ++j) {
      int dh = j * 16 + ln;
      if (mode == 2) {                     // vT16[bth][dh][n] : contiguous in n
        v8h pk;
#pragma unroll
        for (int q = 0; q < 8; ++q) pk[q] = (_Float16)acc[i][j][q];
        *(v8h*)(vT16 + ((size_t)(bt * H_ + h)) * (DH_ * N_) + (size_t)dh * N_ + nb) = pk;
      } else {                             // q16/k16[bth][n][dh]
        _Float16* dst = (mode == 0 ? q16 : k16) +
                        ((size_t)(bt * H_ + h)) * (N_ * DH_) + (size_t)nb * DH_ + dh;
#pragma unroll
        for (int q = 0; q < 8; ++q) dst[(size_t)q * DH_] = (_Float16)acc[i][j][q];
      }
    }
  }
}

// flash attention: one wave = one (b,t,h) x 16-query tile, 32 keys per step
__global__ __launch_bounds__(256) void k_attn(const _Float16* __restrict__ q16,
                                              const _Float16* __restrict__ k16,
                                              const _Float16* __restrict__ vT16,
                                              const unsigned char* __restrict__ send,
                                              const int* __restrict__ anySend,
                                              const int* __restrict__ pad,
                                              _Float16* __restrict__ attn16) {
  __shared__ __align__(16) _Float16 pbuf[8][16 * 32];
  int wave = threadIdx.x >> 5, lane = threadIdx.x & 31;
  int gw  = blockIdx.x * 8 + wave;        // 8192 waves
  int qt  = gw & 63;
  int bth = gw >> 6;
  int h   = bth & 7;
  int bt  = bth >> 3;
  int b   = bt >> 3;
  int ln  = lane & 15, hi = lane >> 4;
  (void)h;

  const _Float16* Qb = q16  + (size_t)bth * (N_ * DH_);
  const _Float16* Kb = k16  + (size_t)bth * (N_ * DH_);
  const _Float16* Vt = vT16 + (size_t)bth * (DH_ * N_);
  const unsigned char* sb = send + (size_t)bt * N_;
  _Float16* pb = pbuf[wave];

  // Q tile as two A-fragments (dh 0..31, 32..63)
  int qrow = qt * 16 + ln;
  const _Float16* qp = Qb + (size_t)qrow * DH_ + hi * 8;
  v16h a0 = ld16(qp, qp + 16);
  v16h a1 = ld16(qp + 32, qp + 48);

  float m[8], l[8];
  v8f o0 = {}, o1 = {}, o2 = {}, o3 = {};
#pragma unroll
  for (int j = 0; j < 8; ++j) { m[j] = NEGV; l[j] = 0.0f; }
  const float scale = 0.125f;             // 1/sqrt(64)

  for (int kt = 0; kt < 32; ++kt) {
    int k0 = kt * 32;
    // logits: two 16-key tiles, each = 2 WMMAs over dh
    const _Float16* kp0 = Kb + (size_t)(k0 + ln) * DH_ + hi * 8;
    const _Float16* kp1 = Kb + (size_t)(k0 + 16 + ln) * DH_ + hi * 8;
    v8f c0 = {}, c1 = {};
    c0 = wmma32(a0, ld16(kp0, kp0 + 16),      c0);
    c0 = wmma32(a1, ld16(kp0 + 32, kp0 + 48), c0);
    c1 = wmma32(a0, ld16(kp1, kp1 + 16),      c1);
    c1 = wmma32(a1, ld16(kp1 + 32, kp1 + 48), c1);

    bool s0 = sb[k0 + ln] != 0;
    bool s1 = sb[k0 + 16 + ln] != 0;

#pragma unroll
    for (int j = 0; j < 8; ++j) {
      float x0 = s0 ? c0[j] * scale : NEGV;
      float x1 = s1 ? c1[j] * scale : NEGV;
      float tm = rmax16(fmaxf(x0, x1));
      float nm = fmaxf(m[j], tm);
      float f  = __expf(m[j] - nm);
      float p0 = s0 ? __expf(x0 - nm) : 0.0f;
      float p1 = s1 ? __expf(x1 - nm) : 0.0f;
      float rs = rsum16(p0 + p1);
      l[j] = l[j] * f + rs;
      m[j] = nm;
      o0[j] *= f; o1[j] *= f; o2[j] *= f; o3[j] *= f;
      int prow = hi * 8 + j;              // C-layout row -> LDS [row][32 keys]
      pb[prow * 32 + ln]      = (_Float16)p0;
      pb[prow * 32 + 16 + ln] = (_Float16)p1;
    }
    asm volatile("" ::: "memory");        // DS in-order per wave; stop reordering
    const _Float16* pp = pb + ln * 32 + hi * 8;
    v16h pa = ld16(pp, pp + 16);          // P as A-fragment (16q x 32k)
    asm volatile("" ::: "memory");

    // PV: V^T rows are contiguous in key -> clean B-fragments
    const _Float16* vp = Vt + (size_t)ln * N_ + k0 + hi * 8;
    o0 = wmma32(pa, ld16(vp + 0 * 16 * N_, vp + 0 * 16 * N_ + 16), o0);
    o1 = wmma32(pa, ld16(vp + 1 * 16 * N_, vp + 1 * 16 * N_ + 16), o1);
    o2 = wmma32(pa, ld16(vp + 2 * 16 * N_, vp + 2 * 16 * N_ + 16), o2);
    o3 = wmma32(pa, ld16(vp + 3 * 16 * N_, vp + 3 * 16 * N_ + 16), o3);
  }

  int any = anySend[bt];
#pragma unroll
  for (int j = 0; j < 8; ++j) {
    int n = qt * 16 + hi * 8 + j;
    bool kq  = any && (pad[(size_t)b * N_ + n] != 0);
    float inv = (kq && l[j] > 0.0f) ? 1.0f / l[j] : 0.0f;
    _Float16* dst = attn16 + ((size_t)bt * N_ + n) * D_ + (bth & 7) * DH_;
    dst[0 * 16 + ln] = (_Float16)(o0[j] * inv);
    dst[1 * 16 + ln] = (_Float16)(o1[j] * inv);
    dst[2 * 16 + ln] = (_Float16)(o2[j] * inv);
    dst[3 * 16 + ln] = (_Float16)(o3[j] * inv);
  }
}

// out = attn @ Wo^T + x  (fp32), 2x4 register blocking: 32 rows x 64 cols / wave
__global__ __launch_bounds__(256) void k_out(const _Float16* __restrict__ attn16,
                                             const _Float16* __restrict__ wo16,
                                             const float* __restrict__ x,
                                             float* __restrict__ out) {
  int wave = threadIdx.x >> 5, lane = threadIdx.x & 31;
  int gw = blockIdx.x * 8 + wave;         // 4096 waves
  int rt = gw >> 3;                       // 0..511 : rows rt*32
  int ct = gw & 7;                        // 0..7   : cols ct*64
  int ln = lane & 15, hi = lane >> 4;

  const _Float16* ar0 = attn16 + (size_t)(rt * 32 + ln) * D_ + hi * 8;
  const _Float16* ar1 = ar0 + (size_t)16 * D_;
  const _Float16* br0 = wo16 + (size_t)(ct * 64 + ln) * D_ + hi * 8;
  const _Float16* br1 = br0 + (size_t)16 * D_;
  const _Float16* br2 = br0 + (size_t)32 * D_;
  const _Float16* br3 = br0 + (size_t)48 * D_;

  v8f acc[2][4] = {};
#pragma unroll
  for (int kk = 0; kk < 16; ++kk) {
    int kb = kk * 32;
    v16h a0 = ld16(ar0 + kb, ar0 + kb + 16);
    v16h a1 = ld16(ar1 + kb, ar1 + kb + 16);
    v16h b0 = ld16(br0 + kb, br0 + kb + 16);
    v16h b1 = ld16(br1 + kb, br1 + kb + 16);
    v16h b2 = ld16(br2 + kb, br2 + kb + 16);
    v16h b3 = ld16(br3 + kb, br3 + kb + 16);
    acc[0][0] = wmma32(a0, b0, acc[0][0]);
    acc[0][1] = wmma32(a0, b1, acc[0][1]);
    acc[0][2] = wmma32(a0, b2, acc[0][2]);
    acc[0][3] = wmma32(a0, b3, acc[0][3]);
    acc[1][0] = wmma32(a1, b0, acc[1][0]);
    acc[1][1] = wmma32(a1, b1, acc[1][1]);
    acc[1][2] = wmma32(a1, b2, acc[1][2]);
    acc[1][3] = wmma32(a1, b3, acc[1][3]);
  }

#pragma unroll
  for (int i = 0; i < 2; ++i) {
#pragma unroll
    for (int j = 0; j < 4; ++j) {
      int o = ct * 64 + j * 16 + ln;
#pragma unroll
      for (int q = 0; q < 8; ++q) {
        size_t idx = (size_t)(rt * 32 + i * 16 + hi * 8 + q) * D_ + o;
        out[idx] = acc[i][j][q] + x[idx];
      }
    }
  }
}

// ---------------------------------------------------------------- launch ----
extern "C" void kernel_launch(void* const* d_in, const int* in_sizes, int n_in,
                              void* d_out, int out_size, void* d_ws, size_t ws_size,
                              hipStream_t stream) {
  const float* x     = (const float*)d_in[0];
  const float* pos   = (const float*)d_in[1];
  const int*   pad   = (const int*)d_in[2];
  const int*   spike = (const int*)d_in[3];
  const float* rms_w = (const float*)d_in[4];
  const float* Wq    = (const float*)d_in[5];
  const float* Wk    = (const float*)d_in[6];
  const float* Wv    = (const float*)d_in[7];
  const float* Wo    = (const float*)d_in[8];
  const float* dirs  = (const float*)d_in[9];
  const float* freqs = (const float*)d_in[10];
  float* out = (float*)d_out;

  char*  ws  = (char*)d_ws;
  size_t off = 0;
  auto alloc = [&](size_t bytes) -> void* {
    void* p = ws + off;
    off += (bytes + 255) & ~(size_t)255;
    return p;
  };
  _Float16* qk16   = (_Float16*)alloc((size_t)ROWS_ * D_ * 2);      // 16 MB
  _Float16* xn16   = (_Float16*)alloc((size_t)ROWS_ * D_ * 2);      // 16 MB
  _Float16* w16    = (_Float16*)alloc((size_t)4 * D_ * D_ * 2);     // 2 MB
  _Float16* q16    = (_Float16*)alloc((size_t)BTH_ * N_ * DH_ * 2); // 16 MB
  _Float16* k16    = (_Float16*)alloc((size_t)BTH_ * N_ * DH_ * 2); // 16 MB
  _Float16* vT16   = (_Float16*)alloc((size_t)BTH_ * DH_ * N_ * 2); // 16 MB
  _Float16* attn16 = (_Float16*)alloc((size_t)ROWS_ * D_ * 2);      // 16 MB
  float*    rope   = (float*)alloc((size_t)B_ * N_ * 64 * 4);       // 512 KB
  unsigned char* send = (unsigned char*)alloc((size_t)B_ * T_ * N_);
  int* anySend = (int*)alloc((size_t)B_ * T_ * sizeof(int));

  k_rope <<<256,   256, 0, stream>>>(pos, dirs, freqs, rope);
  k_send <<<B_*T_, 256, 0, stream>>>(pad, spike, send, anySend);
  k_wconv<<<4096,  256, 0, stream>>>(Wq, Wk, Wv, Wo, w16);
  k_norm <<<ROWS_, 256, 0, stream>>>(x, rms_w, rope, xn16, qk16);
  k_proj <<<1536,  256, 0, stream>>>(qk16, xn16, w16, q16, k16, vT16);
  k_attn <<<1024,  256, 0, stream>>>(q16, k16, vT16, send, anySend, pad, attn16);
  k_out  <<<512,   256, 0, stream>>>(attn16, w16 + (size_t)3 * D_ * D_, x, out);
}
